// HDRGenerator_66434554134574
// MI455X (gfx1250) — compile-verified, matrix operand order checked by reading, not run on hardware
//
#include <hip/hip_runtime.h>
#include <hip/hip_bf16.h>

typedef __attribute__((ext_vector_type(16))) __bf16 v16bf;
typedef __attribute__((ext_vector_type(8)))  float  v8f;
typedef unsigned short u16;
typedef unsigned int   u32;
typedef __attribute__((ext_vector_type(8))) u16 v8u16;   // 16-byte segment

#define NPIX 9216    // 96*96
#define W96  96
#define PW   98      // padded width
#define PADN (PW*PW) // 9604 padded pixels
#define FD   64
#define CHK  256     // key chunk for flash attention (40KB LDS)

__device__ __forceinline__ u16 f2bf(float f) {
    u32 u = __float_as_uint(f);
    u32 r = (u + 0x7FFFu + ((u >> 16) & 1u)) >> 16;   // round-to-nearest-even
    return (u16)r;
}
__device__ __forceinline__ float bf2f(u16 h) {
    return __uint_as_float(((u32)h) << 16);
}

// Fragment: 16-bit A/B tile data for one lane = two contiguous 16B segments:
//   K = [kh .. kh+7] and [16+kh .. 16+kh+7], kh = 8*(lane>=16)
union ABf { v16bf v; u16 s[16]; v8u16 h[2]; };

// ---------------- 3->64 3x3 conv + ReLU -> padded pixel-major bf16 ----------
__global__ void hdr_conv_in_relu(const float* __restrict__ x,  // [3][96][96]
                                 const float* __restrict__ w,  // [64][3][3][3]
                                 const float* __restrict__ b,  // [64]
                                 u16* __restrict__ outPad)     // [9604][64]
{
    int idx = blockIdx.x * blockDim.x + threadIdx.x;
    if (idx >= PADN * 8) return;
    int g = idx & 7, pp = idx >> 3;
    int py = pp / PW, px = pp % PW;
    v8u16 r;
    if (py == 0 || py == PW - 1 || px == 0 || px == PW - 1) {
#pragma unroll
        for (int v = 0; v < 8; ++v) r[v] = 0;
    } else {
        int y = py - 1, xx = px - 1;
        float xv[27];
#pragma unroll
        for (int ic = 0; ic < 3; ++ic)
#pragma unroll
            for (int t = 0; t < 9; ++t) {
                int dy = t / 3 - 1, dx = t % 3 - 1;
                int yy = y + dy, xc = xx + dx;
                xv[ic * 9 + t] = (yy >= 0 && yy < W96 && xc >= 0 && xc < W96)
                                     ? x[ic * NPIX + yy * W96 + xc] : 0.f;
            }
#pragma unroll
        for (int v = 0; v < 8; ++v) {
            int oc = g * 8 + v;
            float acc = b[oc];
#pragma unroll
            for (int ic = 0; ic < 3; ++ic)
#pragma unroll
                for (int t = 0; t < 9; ++t)
                    acc += w[(oc * 3 + ic) * 9 + t] * xv[ic * 9 + t];
            r[v] = f2bf(fmaxf(acc, 0.f));
        }
    }
    *(v8u16*)(outPad + pp * FD + g * 8) = r;
}

// -------- repack [64][64][3][3] f32 OIHW -> [64][576] bf16, k = s*64+ic -----
__global__ void hdr_repack_w(const float* __restrict__ w, u16* __restrict__ wp)
{
    int idx = blockIdx.x * blockDim.x + threadIdx.x;
    if (idx >= FD * 576) return;
    int oc = idx / 576, k = idx % 576;
    int s = k >> 6, ic = k & 63;
    wp[idx] = f2bf(w[(oc * FD + ic) * 9 + s]);
}

// -------- 64->64 3x3 conv as implicit GEMM via v_wmma_f32_16x16x32_bf16 -----
__global__ __launch_bounds__(256)
void hdr_conv64_wmma(const u16* __restrict__ inPad,  // [9604][64] bf16
                     const u16* __restrict__ wp,     // [64][576]  bf16
                     const float* __restrict__ bias, // [64]
                     float* __restrict__ outf,       // [64][9216] cm, or null
                     u16* __restrict__ outPad,       // [9604][64] interior, or null
                     int relu)
{
    int lane = threadIdx.x & 31;
    int wave = threadIdx.x >> 5;
    int oc0  = (wave & 3) << 4;
    int pt   = blockIdx.x * 2 + (wave >> 2);   // 0..575
    int p0   = pt << 4;
    int y    = p0 / W96, x0 = p0 % W96;        // 96 % 16 == 0: tile stays in row
    int n    = lane & 15;
    int px   = x0 + n;                         // this lane's pixel (B column)
    int kh   = (lane >> 4) << 3;               // 0 or 8

    const u16* a0 = wp + (u32)(oc0 + n) * 576 + kh;
    const u16* b0 = inPad + ((u32)(y + 1) * PW + (px + 1)) * FD + kh;

    v8f acc = {};
    ABf A, B;
#pragma unroll
    for (int s = 0; s < 9; ++s) {
        const int dy = s / 3 - 1, dx = s % 3 - 1;
        const u16* tap = b0 + (dy * PW + dx) * FD;
#pragma unroll
        for (int hlf = 0; hlf < 2; ++hlf) {
            const int kc = s * 64 + hlf * 32;
            A.h[0] = *(const v8u16*)(a0 + kc);
            A.h[1] = *(const v8u16*)(a0 + kc + 16);
            B.h[0] = *(const v8u16*)(tap + hlf * 32);
            B.h[1] = *(const v8u16*)(tap + hlf * 32 + 16);
            acc = __builtin_amdgcn_wmma_f32_16x16x32_bf16(
                      false, A.v, false, B.v, (short)0, acc, false, false);
        }
    }
    int p    = y * W96 + px;
    int ochi = oc0 + ((lane >> 4) << 3);       // 8 consecutive oc for this lane
    v8u16 rb;
#pragma unroll
    for (int v = 0; v < 8; ++v) {
        float r = acc[v] + bias[ochi + v];
        if (relu) r = fmaxf(r, 0.f);
        rb[v] = f2bf(r);
        if (outf) outf[(ochi + v) * NPIX + p] = r;
    }
    if (outPad)
        *(v8u16*)(outPad + ((u32)(y + 1) * PW + (px + 1)) * FD + ochi) = rb;
}

// ---------------- 1x1 projections from pixel-major bf16 features ------------
__global__ void hdr_proj1x1(const u16* __restrict__ inPad,  // [9604][64] bf16
                            const float* __restrict__ w,    // [rows][64]
                            const float* __restrict__ b,    // [rows]
                            float* __restrict__ outf,       // [rows][9216] or null
                            u16* __restrict__ outb,         // [rows][9216] or null
                            int rows)
{
    int idx = blockIdx.x * blockDim.x + threadIdx.x;
    if (idx >= rows * NPIX) return;
    int r = idx / NPIX, p = idx % NPIX;
    int y = p / W96, x = p % W96;
    const u16* in = inPad + ((u32)(y + 1) * PW + (x + 1)) * FD;
    const float* wr = w + r * FD;
    float acc = b[r];
#pragma unroll
    for (int c8 = 0; c8 < 8; ++c8) {
        v8u16 hv = *(const v8u16*)(in + c8 * 8);
#pragma unroll
        for (int j = 0; j < 8; ++j) acc += wr[c8 * 8 + j] * bf2f(hv[j]);
    }
    if (outf) outf[idx] = acc;
    if (outb) outb[idx] = f2bf(acc);
}

// ------- flash-style pca: out = softmax(Q^T K) V^T + ref, WMMA on P*V -------
// BOTH pca calls in one launch: blocks [0,72) -> (K1,V1,out1), [72,144) ->
// (K3,V3,out3). 8 waves/block, each wave owns a 16-query tile.
__global__ __launch_bounds__(256)
void hdr_flash_pca2(const float* __restrict__ Qf,    // [8][N] (shared)
                    const float* __restrict__ Kf0, const float* __restrict__ Kf1,
                    const u16*  __restrict__ Vb0,  const u16*  __restrict__ Vb1,
                    const float* __restrict__ ref,   // [64][N] (shared, f2)
                    float* __restrict__ out0, float* __restrict__ out1)
{
    __shared__ float Kt[CHK][8];       // transposed: 8 channels contiguous/key
    __shared__ u16   Vc[FD * CHK];

    int side = (blockIdx.x >= 72) ? 1 : 0;
    int bi   = blockIdx.x - side * 72;
    const float* Kf  = side ? Kf1 : Kf0;
    const u16*   Vb  = side ? Vb1 : Vb0;
    float*       outf = side ? out1 : out0;

    int lane = threadIdx.x & 31;
    int wave = threadIdx.x >> 5;
    int q0   = (bi * 8 + wave) << 4;
    int mrow = q0 + (lane & 15);
    int kh   = (lane >> 4) << 3;

    float q[8];
#pragma unroll
    for (int c = 0; c < 8; ++c) q[c] = Qf[c * NPIX + mrow];

    // ---- pass 1: online row max / sum (lanes 0-15 keys 0..127, 16-31 rest)
    float mmax = -3.0e38f, lsum = 0.f;
    int jbase = (lane >> 4) * (CHK / 2);
    for (int c0 = 0; c0 < NPIX; c0 += CHK) {
        __syncthreads();
        for (int t = threadIdx.x; t < CHK * 8; t += 256)
            Kt[t >> 3][t & 7] = Kf[(t & 7) * NPIX + c0 + (t >> 3)];
        __syncthreads();
#pragma unroll 4
        for (int j = 0; j < CHK / 2; ++j) {
            const float* kp = Kt[jbase + j];
            float s = 0.f;
#pragma unroll
            for (int c = 0; c < 8; ++c) s += q[c] * kp[c];
            if (s > mmax) { lsum = lsum * __expf(mmax - s) + 1.0f; mmax = s; }
            else          { lsum += __expf(s - mmax); }
        }
    }
    // combine the two half-row partials (lane ^ 16)
    float mo = __shfl_xor(mmax, 16, 32);
    float lo = __shfl_xor(lsum, 16, 32);
    float mn = fmaxf(mmax, mo);
    lsum = lsum * __expf(mmax - mn) + lo * __expf(mo - mn);
    mmax = mn;
    float invl = 1.0f / lsum;

    // ---- pass 2: P (bf16, A-layout) x V (bf16, B-layout), f32 accumulate
    v8f acc[4] = {{}, {}, {}, {}};
    ABf A, B;
    for (int c0 = 0; c0 < NPIX; c0 += CHK) {
        __syncthreads();
        for (int t = threadIdx.x; t < CHK * 8; t += 256)
            Kt[t >> 3][t & 7] = Kf[(t & 7) * NPIX + c0 + (t >> 3)];
        for (int t = threadIdx.x; t < FD * CHK / 8; t += 256) {
            int row = t >> 5, seg = t & 31;   // CHK/8 = 32 segments per row
            *(v8u16*)&Vc[row * CHK + seg * 8] =
                *(const v8u16*)(Vb + row * NPIX + c0 + seg * 8);
        }
        __syncthreads();
        for (int sub = 0; sub < CHK / 32; ++sub) {
            int k0 = sub << 5;
#pragma unroll
            for (int e = 0; e < 8; ++e) {       // low 16B segment: K = k0+kh+e
                const float* kp = Kt[k0 + kh + e];
                float s = 0.f;
#pragma unroll
                for (int c = 0; c < 8; ++c) s += q[c] * kp[c];
                A.s[e] = f2bf(__expf(s - mmax) * invl);
            }
#pragma unroll
            for (int e = 0; e < 8; ++e) {       // high segment: K = k0+16+kh+e
                const float* kp = Kt[k0 + 16 + kh + e];
                float s = 0.f;
#pragma unroll
                for (int c = 0; c < 8; ++c) s += q[c] * kp[c];
                A.s[8 + e] = f2bf(__expf(s - mmax) * invl);
            }
#pragma unroll
            for (int ct = 0; ct < 4; ++ct) {
                const u16* vr = &Vc[((ct << 4) + (lane & 15)) * CHK + k0 + kh];
                B.h[0] = *(const v8u16*)vr;
                B.h[1] = *(const v8u16*)(vr + 16);
                acc[ct] = __builtin_amdgcn_wmma_f32_16x16x32_bf16(
                              false, A.v, false, B.v, (short)0, acc[ct], false, false);
            }
        }
    }
#pragma unroll
    for (int ct = 0; ct < 4; ++ct) {
        int ch = (ct << 4) + (lane & 15);
#pragma unroll
        for (int v = 0; v < 8; ++v) {
            int qq = q0 + v + ((lane >> 4) << 3);
            outf[ch * NPIX + qq] = acc[ct][v] + ref[ch * NPIX + qq];
        }
    }
}

// -------- fused = (a1 + f2 + a3)/3 -> padded pixel-major bf16 (with borders)
__global__ void hdr_fuse_avg(const float* __restrict__ a1,
                             const float* __restrict__ f2,
                             const float* __restrict__ a3,
                             u16* __restrict__ outPad)       // [9604][64]
{
    int idx = blockIdx.x * blockDim.x + threadIdx.x;
    if (idx >= PADN * 8) return;
    int g = idx & 7, pp = idx >> 3;
    int py = pp / PW, px = pp % PW;
    v8u16 r;
    if (py == 0 || py == PW - 1 || px == 0 || px == PW - 1) {
#pragma unroll
        for (int v = 0; v < 8; ++v) r[v] = 0;
    } else {
        int p = (py - 1) * W96 + (px - 1);
#pragma unroll
        for (int v = 0; v < 8; ++v) {
            int o = (g * 8 + v) * NPIX + p;
            r[v] = f2bf((a1[o] + f2[o] + a3[o]) * (1.0f / 3.0f));
        }
    }
    *(v8u16*)(outPad + pp * FD + g * 8) = r;
}

// ---------------- zero the halo of a padded buffer --------------------------
__global__ void hdr_zero_border(u16* __restrict__ pad)
{
    int idx = blockIdx.x * blockDim.x + threadIdx.x;
    if (idx >= PADN * 8) return;
    int g = idx & 7, pp = idx >> 3;
    int py = pp / PW, px = pp % PW;
    if (py == 0 || py == PW - 1 || px == 0 || px == PW - 1) {
        v8u16 z;
#pragma unroll
        for (int v = 0; v < 8; ++v) z[v] = 0;
        *(v8u16*)(pad + pp * FD + g * 8) = z;
    }
}

// ---------------- 64->3 3x3 conv (VALU, tiny) -> f32 output -----------------
__global__ void hdr_conv_out(const u16* __restrict__ hPad,  // [9604][64] bf16
                             const float* __restrict__ w,   // [3][64][3][3]
                             const float* __restrict__ b,   // [3]
                             float* __restrict__ out)       // [3][9216]
{
    int idx = blockIdx.x * blockDim.x + threadIdx.x;
    if (idx >= 3 * NPIX) return;
    int oc = idx / NPIX, p = idx % NPIX;
    int y = p / W96, x = p % W96;
    const u16* base = hPad + ((u32)(y + 1) * PW + (x + 1)) * FD;
    float acc = b[oc];
#pragma unroll
    for (int s = 0; s < 9; ++s) {
        const int dy = s / 3 - 1, dx = s % 3 - 1;
        const u16* tap = base + (dy * PW + dx) * FD;
#pragma unroll
        for (int c8 = 0; c8 < 8; ++c8) {
            v8u16 hv = *(const v8u16*)(tap + c8 * 8);
#pragma unroll
            for (int j = 0; j < 8; ++j) {
                int ic = c8 * 8 + j;
                acc += w[(oc * FD + ic) * 9 + s] * bf2f(hv[j]);
            }
        }
    }
    out[idx] = acc;
}

extern "C" void kernel_launch(void* const* d_in, const int* in_sizes, int n_in,
                              void* d_out, int out_size, void* d_ws, size_t ws_size,
                              hipStream_t stream) {
    (void)in_sizes; (void)n_in; (void)out_size; (void)ws_size;
    const float* l[3]   = {(const float*)d_in[0], (const float*)d_in[1], (const float*)d_in[2]};
    const float* ext_w1 = (const float*)d_in[3];
    const float* ext_b1 = (const float*)d_in[4];
    const float* ext_w2 = (const float*)d_in[5];
    const float* ext_b2 = (const float*)d_in[6];
    const float* q_w = (const float*)d_in[7],  *q_b = (const float*)d_in[8];
    const float* k_w = (const float*)d_in[9],  *k_b = (const float*)d_in[10];
    const float* v_w = (const float*)d_in[11], *v_b = (const float*)d_in[12];
    const float* fus_w1 = (const float*)d_in[13], *fus_b1 = (const float*)d_in[14];
    const float* fus_w2 = (const float*)d_in[15], *fus_b2 = (const float*)d_in[16];

    const size_t PADB = (size_t)PADN * FD * 2;  // padded pixel-major bf16
    const size_t F32  = (size_t)FD * NPIX * 4;  // channel-major f32
    const size_t B16  = (size_t)FD * NPIX * 2;  // channel-major bf16
    const size_t R32  = (size_t)8 * NPIX * 4;
    char* ws = (char*)d_ws;
    size_t off = 0;
    auto align256 = [&]() { off = (off + 255) & ~(size_t)255; };
    u16* s1Pad[3]; for (int i = 0; i < 3; ++i) { s1Pad[i] = (u16*)(ws + off); off += PADB; align256(); }
    u16* s2Pad[3]; for (int i = 0; i < 3; ++i) { s2Pad[i] = (u16*)(ws + off); off += PADB; align256(); }
    float* f2f32 = (float*)(ws + off); off += F32;  align256();
    float* Qf  = (float*)(ws + off); off += R32;  align256();
    float* K1  = (float*)(ws + off); off += R32;  align256();
    float* K3  = (float*)(ws + off); off += R32;  align256();
    u16*   V1b = (u16*)(ws + off);   off += B16;  align256();
    u16*   V3b = (u16*)(ws + off);   off += B16;  align256();
    float* a1  = (float*)(ws + off); off += F32;  align256();
    float* a3  = (float*)(ws + off); off += F32;  align256();
    u16* fusedPad = (u16*)(ws + off); off += PADB; align256();
    u16* hPad     = (u16*)(ws + off); off += PADB; align256();
    u16* wp_ext = (u16*)(ws + off); off += (size_t)FD * 576 * 2; align256();
    u16* wp_fus = (u16*)(ws + off); off += (size_t)FD * 576 * 2; align256();

    const int TPB = 256;
    int gPad8 = (PADN * 8 + TPB - 1) / TPB;

    // 1) feature extractor stage 1 (3->64) -> padded pixel-major bf16
    for (int i = 0; i < 3; ++i)
        hipLaunchKernelGGL(hdr_conv_in_relu, dim3(gPad8), dim3(TPB), 0, stream,
                           l[i], ext_w1, ext_b1, s1Pad[i]);
    // 2) weight repacks
    int gW = (FD * 576 + TPB - 1) / TPB;
    hipLaunchKernelGGL(hdr_repack_w, dim3(gW), dim3(TPB), 0, stream, ext_w2, wp_ext);
    hipLaunchKernelGGL(hdr_repack_w, dim3(gW), dim3(TPB), 0, stream, fus_w1, wp_fus);
    // 3) feature extractor stage 2 (64->64, WMMA); f32 copy only needed for f2
    for (int i = 0; i < 3; ++i)
        hipLaunchKernelGGL(hdr_conv64_wmma, dim3(288), dim3(TPB), 0, stream,
                           s1Pad[i], wp_ext, ext_b2,
                           (i == 1) ? f2f32 : (float*)nullptr, s2Pad[i], 1);
    // 4) Q/K/V projections (read pixel-major bf16 features)
    int gR8 = (8 * NPIX + TPB - 1) / TPB;
    int gR64 = (FD * NPIX + TPB - 1) / TPB;
    hipLaunchKernelGGL(hdr_proj1x1, dim3(gR8), dim3(TPB), 0, stream, s2Pad[1], q_w, q_b, Qf, (u16*)nullptr, 8);
    hipLaunchKernelGGL(hdr_proj1x1, dim3(gR8), dim3(TPB), 0, stream, s2Pad[0], k_w, k_b, K1, (u16*)nullptr, 8);
    hipLaunchKernelGGL(hdr_proj1x1, dim3(gR8), dim3(TPB), 0, stream, s2Pad[2], k_w, k_b, K3, (u16*)nullptr, 8);
    hipLaunchKernelGGL(hdr_proj1x1, dim3(gR64), dim3(TPB), 0, stream, s2Pad[0], v_w, v_b, (float*)nullptr, V1b, FD);
    hipLaunchKernelGGL(hdr_proj1x1, dim3(gR64), dim3(TPB), 0, stream, s2Pad[2], v_w, v_b, (float*)nullptr, V3b, FD);
    // 5) both flash attention pcas in ONE launch (144 blocks), fused residual
    hipLaunchKernelGGL(hdr_flash_pca2, dim3(144), dim3(TPB), 0, stream,
                       Qf, K1, K3, V1b, V3b, f2f32, a1, a3);
    // 6) average -> padded pixel-major bf16 (writes its own zero halo)
    hipLaunchKernelGGL(hdr_fuse_avg, dim3(gPad8), dim3(TPB), 0, stream, a1, f2f32, a3, fusedPad);
    // 7) fusion conv 1 (64->64, WMMA) -> hPad interior; halo zeroed separately
    hipLaunchKernelGGL(hdr_zero_border, dim3(gPad8), dim3(TPB), 0, stream, hPad);
    hipLaunchKernelGGL(hdr_conv64_wmma, dim3(288), dim3(TPB), 0, stream,
                       fusedPad, wp_fus, fus_b1, (float*)nullptr, hPad, 1);
    // 8) fusion conv 2 (64->3) -> output
    int gOut = (3 * NPIX + TPB - 1) / TPB;
    hipLaunchKernelGGL(hdr_conv_out, dim3(gOut), dim3(TPB), 0, stream,
                       hPad, fus_w2, fus_b2, (float*)d_out);
}
